// MPLayer_55173149885005
// MI455X (gfx1250) — matrix operand order repaired and checked by reading, not compiled, not run on hardware
//
#include <hip/hip_runtime.h>
#include <hip/hip_bf16.h>

// CDNA5 wave32 WMMA types
typedef __attribute__((ext_vector_type(16))) _Float16 v16h;
typedef __attribute__((ext_vector_type(8)))  float    v8f;
typedef __attribute__((ext_vector_type(4)))  unsigned v4u;

#define LDSAS __attribute__((address_space(3)))
typedef LDSAS _Float16       lds_half;
typedef LDSAS const _Float16 lds_chalf;
typedef LDSAS const v4u      lds_cv4u;
typedef LDSAS v4u            lds_v4u;

// leaky_relu(t, 0.2) == 0.6*t + 0.4*|t|   (max(t,0)+0.2*min(t,0))
__device__ __forceinline__ float lrelu(float t) {
  return fmaf(0.4f, fabsf(t), 0.6f * t);
}

// ---------------------------------------------------------------------------
// A-fragment (16x32 f16, MxK) from row-major LDS.
// CDNA5 ISA: lane L -> M = L%16, half = L/16;
// VGPR p: p<4 -> K = half*8 + 2p ; p>=4 -> K = 16 + half*8 + 2(p-4).
// Relative to (row + half*8 halfs) that is dwords {0..3} and {8..11}
// -> two ds_load_b128.  Strides chosen with (stride/2) == 4 (mod 8) words:
// 16-byte aligned rows whose b128 accesses tile all 64 LDS banks.
// ---------------------------------------------------------------------------
__device__ __forceinline__ v16h a_frag_lds(lds_chalf* base, int stride, int lane) {
  const int m = lane & 15, half = lane >> 4;
  union { v16h v; v4u q[2]; } r;
  lds_cv4u* p = (lds_cv4u*)(base + m * stride + half * 8);
  r.q[0] = p[0];   // K = half*8 + 0..7
  r.q[1] = p[2];   // K = 16 + half*8 + 0..7
  return r.v;
}

// ---------------------------------------------------------------------------
// B-fragment (32x16 f16, KxN) from pre-packed weights Wt[n][kpair]:
// dword = (f16(W[2k+1][n])<<16) | f16(W[2k][n]), K zero-padded to 32x.
// ISA: lane L -> N = L%16, khalf = L/16; VGPR p: p<4 -> K = 32*ks + khalf*8 + 2p;
// p>=4 -> K = 32*ks + 16 + khalf*8 + 2(p-4).  Two global_load_b128 of row n.
// ---------------------------------------------------------------------------
__device__ __forceinline__ v16h b_frag_pk(const v4u* __restrict__ Wt, int Kpad8,
                                          int n, int ks, int lane) {
  const int khalf = lane >> 4;
  union { v16h v; v4u q[2]; } r;
  const v4u* p = Wt + (size_t)n * Kpad8 + ks * 4 + khalf;
  r.q[0] = p[0];
  r.q[1] = p[2];
  return r.v;
}

// ---------------------------------------------------------------------------
// One MLP layer on WMMA:  out = lrelu(sIn[MT*16 x K] @ W[K x N] (+ b))
// N-tiles strided over the 4 waves; K looped per 32.
// AGG=false: store f16 activations to sOut.  AGG=true: reduce rows m < 75
// into sAgg[n].  ADD_BIAS=false when the bias was folded into a padded
// K-row of the packed weights (activation column K holds 1.0).
// D layout: lane holds N = lane%16, M = (lane/16)*8 + p.
// ---------------------------------------------------------------------------
template <int MT, bool AGG, bool ADD_BIAS>
__device__ void mlp_layer_wmma(lds_chalf* sIn, int strideIn,
                               const v4u* __restrict__ Wt, int Kpad8,
                               const float* __restrict__ bias,
                               int ksteps, int N,
                               lds_half* sOut, int strideOut,
                               float* sAgg,
                               int lane, int wave) {
  const int ntiles = N >> 4;
  for (int nt = wave; nt < ntiles; nt += 4) {
    const int n = nt * 16 + (lane & 15);
    v8f acc[MT] = {};
    for (int ks = 0; ks < ksteps; ++ks) {
      const v16h b = b_frag_pk(Wt, Kpad8, n, ks, lane);
#pragma unroll
      for (int mt = 0; mt < MT; ++mt) {
        const v16h a = a_frag_lds(sIn + mt * 16 * strideIn + ks * 32, strideIn, lane);
        acc[mt] = __builtin_amdgcn_wmma_f32_16x16x32_f16(
            false, a, false, b, (short)0, acc[mt], false, false);
      }
    }
    const float bn = ADD_BIAS ? bias[n] : 0.f;
    if constexpr (!AGG) {
#pragma unroll
      for (int mt = 0; mt < MT; ++mt) {
#pragma unroll
        for (int p = 0; p < 8; ++p) {
          const int m = mt * 16 + (lane >> 4) * 8 + p;
          const float t = ADD_BIAS ? acc[mt][p] + bn : acc[mt][p];
          sOut[m * strideOut + n] = (_Float16)lrelu(t);
        }
      }
    } else {
      float s = 0.f;
#pragma unroll
      for (int mt = 0; mt < MT; ++mt) {
#pragma unroll
        for (int p = 0; p < 8; ++p) {
          const int m = mt * 16 + (lane >> 4) * 8 + p;   // mask pad rows >= 75
          const float v = lrelu(acc[mt][p] + bn);
          s += (m < 75) ? v : 0.f;
        }
      }
      atomicAdd(&sAgg[n], s);   // only 2 lanes share each n
    }
  }
}

// ---------------------------------------------------------------------------
// Weight pre-pack: Wt[n][kpair] = pack_f16(W[2kp][n], W[2kp+1][n]);
// K zero-padded to Kpad, N zero-padded to Npad.  If bias != nullptr it is
// stored in K-row K (first pad row) so the GEMM adds it for free.
// Re-run every launch (~90k dwords total) -> deterministic.
// ---------------------------------------------------------------------------
__global__ void pack_w_kernel(const float* __restrict__ W, const float* __restrict__ bias,
                              unsigned* __restrict__ Wt,
                              int K, int Kpad2 /*Kpad/2*/, int N, int Npad) {
  const int idx = blockIdx.x * 256 + threadIdx.x;
  if (idx >= Npad * Kpad2) return;
  const int n  = idx / Kpad2;
  const int kp = idx % Kpad2;
  const int k0 = kp * 2;
  float f0 = 0.f, f1 = 0.f;
  if (n < N) {
    f0 = (k0     < K) ? W[(size_t)k0       * N + n] : ((bias && k0     == K) ? bias[n] : 0.f);
    f1 = (k0 + 1 < K) ? W[(size_t)(k0 + 1) * N + n] : ((bias && k0 + 1 == K) ? bias[n] : 0.f);
  }
  const unsigned lo = (unsigned)__builtin_bit_cast(unsigned short, (_Float16)f0);
  const unsigned hi = (unsigned)__builtin_bit_cast(unsigned short, (_Float16)f1);
  Wt[idx] = lo | (hi << 16);
}

// ---------------------------------------------------------------------------
// Kernel 1: one workgroup per (b, i).  Builds the 75 edge rows
// [x_i(3) | x_j(3) | dist | 1.0], runs the 3-layer edge MLP on WMMA, and
// sum-aggregates over j into aggOut[(b*75+i)*192 + n].
// ---------------------------------------------------------------------------
__global__ __launch_bounds__(128) void mp_edge_kernel(
    const float* __restrict__ x,
    const v4u* __restrict__ w1p,
    const v4u* __restrict__ w2p, const float* __restrict__ b2,
    const v4u* __restrict__ w3p, const float* __restrict__ b3,
    float* __restrict__ aggOut) {
  __shared__ __attribute__((aligned(16))) _Float16 sA0[80 * 40];   //  6.4 KB (K 7->32, +pad)
  __shared__ __attribute__((aligned(16))) _Float16 sH1[80 * 104];  // 16.6 KB (96 cols)
  __shared__ __attribute__((aligned(16))) _Float16 sH2[80 * 168];  // 26.9 KB (160 cols)
  __shared__ float sAgg[192];

  lds_half* A0 = (lds_half*)sA0;
  lds_half* H1 = (lds_half*)sH1;
  lds_half* H2 = (lds_half*)sH2;

  const int tid = threadIdx.x;
  const int lane = tid & 31, wave = tid >> 5;
  const int bi = blockIdx.x;          // b*75 + i
  const int b  = bi / 75;
  const float* __restrict__ xi = x + (size_t)bi * 3;
  const float* __restrict__ xb = x + (size_t)b * 75 * 3;

  {
    lds_v4u* z = (lds_v4u*)A0;                          // b128 zero-fill
    for (int idx = tid; idx < (80 * 40) / 8; idx += 128) z[idx] = v4u{0, 0, 0, 0};
  }
  for (int n = tid; n < 192; n += 128) sAgg[n] = 0.f;
  __syncthreads();

  if (tid < 75) {
    const int j = tid;
    const float xi0 = xi[0], xi1 = xi[1], xi2 = xi[2];
    const float xj0 = xb[j * 3 + 0], xj1 = xb[j * 3 + 1], xj2 = xb[j * 3 + 2];
    // reference: norm(diffs + 1e-12) -> 1e-12 added per component
    const float d0 = xj0 - xi0 + 1e-12f;
    const float d1 = xj1 - xi1 + 1e-12f;
    const float d2 = xj2 - xi2 + 1e-12f;
    const float dist = sqrtf(d0 * d0 + d1 * d1 + d2 * d2);
    lds_half* row = A0 + j * 40;
    row[0] = (_Float16)xi0; row[1] = (_Float16)xi1; row[2] = (_Float16)xi2;
    row[3] = (_Float16)xj0; row[4] = (_Float16)xj1; row[5] = (_Float16)xj2;
    row[6] = (_Float16)dist;
    row[7] = (_Float16)1.f;            // bias column (fe_b1 lives in W1 row 7)
  }
  __syncthreads();

  //                              in  ldi  Wt  Kp/8  bias  ks  N    out  ldo
  mlp_layer_wmma<5, false, false>(A0, 40,  w1p,  4, nullptr, 1,  96, H1, 104, nullptr, lane, wave);
  __syncthreads();
  mlp_layer_wmma<5, false, true >(H1, 104, w2p, 12, b2,      3, 160, H2, 168, nullptr, lane, wave);
  __syncthreads();
  mlp_layer_wmma<5, true,  true >(H2, 168, w3p, 20, b3,      5, 192, nullptr, 0, sAgg, lane, wave);
  __syncthreads();

  for (int n = tid; n < 192; n += 128)
    aggOut[(size_t)bi * 192 + n] = sAgg[n];
}

// ---------------------------------------------------------------------------
// Kernel 2: node MLP.  One workgroup per 16 (b,i) rows (600 WGs).
// Y = [agg(192) | x(3) | 1.0 | 0-pad] (195 -> 224, bias folded);
// layers 1-2 on WMMA; final 256->3 also on WMMA (N padded 3->16, wave 0).
// ---------------------------------------------------------------------------
__global__ __launch_bounds__(128) void mp_node_kernel(
    const float* __restrict__ agg, const float* __restrict__ x,
    const v4u* __restrict__ w1p,
    const v4u* __restrict__ w2p, const float* __restrict__ b2,
    const v4u* __restrict__ w3p, const float* __restrict__ b3,
    float* __restrict__ out) {
  __shared__ __attribute__((aligned(16))) _Float16 sY [16 * 232]; // 195 -> pad 224
  __shared__ __attribute__((aligned(16))) _Float16 sN1[16 * 264]; // 256 cols
  __shared__ __attribute__((aligned(16))) _Float16 sN2[16 * 264];

  lds_half* Y  = (lds_half*)sY;
  lds_half* N1 = (lds_half*)sN1;
  lds_half* N2 = (lds_half*)sN2;

  const int tid = threadIdx.x, lane = tid & 31, wave = tid >> 5;
  const int r0 = blockIdx.x * 16;

  for (int idx = tid; idx < 16 * 232; idx += 128) {
    const int r = idx / 232, c = idx % 232;
    const int rg = r0 + r;
    float v = 0.f;
    if (c < 192)       v = agg[(size_t)rg * 192 + c];
    else if (c < 195)  v = x[(size_t)rg * 3 + (c - 192)];
    else if (c == 195) v = 1.f;        // bias column (fn_b1 lives in W1 row 195)
    Y[r * 232 + c] = (_Float16)v;
  }
  __syncthreads();

  mlp_layer_wmma<1, false, false>(Y,  232, w1p, 28, nullptr, 7, 256, N1, 264, nullptr, lane, wave);
  __syncthreads();
  mlp_layer_wmma<1, false, true >(N1, 264, w2p, 32, b2,      8, 256, N2, 264, nullptr, lane, wave);
  __syncthreads();

  if (wave == 0) {                     // final linear 256->3 (N padded to 16)
    const int n = lane & 15;
    v8f acc = {};
    for (int ks = 0; ks < 8; ++ks) {
      const v16h bfr = b_frag_pk(w3p, 32, n, ks, lane);
      const v16h afr = a_frag_lds(N2 + ks * 32, 264, lane);
      acc = __builtin_amdgcn_wmma_f32_16x16x32_f16(
          false, afr, false, bfr, (short)0, acc, false, false);
    }
    if (n < 3) {
      const float bn = b3[n];
#pragma unroll
      for (int p = 0; p < 8; ++p) {
        const int m = (lane >> 4) * 8 + p;
        out[(size_t)(r0 + m) * 3 + n] = acc[p] + bn;
      }
    }
  }
}

// ---------------------------------------------------------------------------
// Inputs (setup_inputs order): x, fe_w1, fe_b1, fe_w2, fe_b2, fe_w3, fe_b3,
//                              fn_w1, fn_b1, fn_w2, fn_b2, fn_w3, fn_b3
// Workspace: [agg 9600x192 f32 | packed f16 weights], ~7.73 MB.
// ---------------------------------------------------------------------------
extern "C" void kernel_launch(void* const* d_in, const int* in_sizes, int n_in,
                              void* d_out, int out_size, void* d_ws, size_t ws_size,
                              hipStream_t stream) {
  const float* x     = (const float*)d_in[0];
  const float* fe_w1 = (const float*)d_in[1];
  const float* fe_b1 = (const float*)d_in[2];
  const float* fe_w2 = (const float*)d_in[3];
  const float* fe_b2 = (const float*)d_in[4];
  const float* fe_w3 = (const float*)d_in[5];
  const float* fe_b3 = (const float*)d_in[6];
  const float* fn_w1 = (const float*)d_in[7];
  const float* fn_b1 = (const float*)d_in[8];
  const float* fn_w2 = (const float*)d_in[9];
  const float* fn_b2 = (const float*)d_in[10];
  const float* fn_w3 = (const float*)d_in[11];
  const float* fn_b3 = (const float*)d_in[12];
  (void)in_sizes; (void)n_in; (void)out_size; (void)ws_size;

  // ---- workspace layout (dwords), everything 16-byte aligned ----
  float*    agg = (float*)d_ws;                        // 9600*192 f32
  unsigned* w1p = (unsigned*)d_ws + 9600 * 192;        //  16*96
  unsigned* w2p = w1p + 16 * 96;                       //  48*160
  unsigned* w3p = w2p + 48 * 160;                      //  80*192
  unsigned* n1p = w3p + 80 * 192;                      // 112*256
  unsigned* n2p = n1p + 112 * 256;                     // 128*256
  unsigned* n3p = n2p + 128 * 256;                     // 128*16

  auto pack = [&](const float* W, const float* bias, unsigned* Wt,
                  int K, int Kpad, int N, int Npad) {
    const int total = Npad * (Kpad >> 1);
    pack_w_kernel<<<(total + 255) / 256, 256, 0, stream>>>(W, bias, Wt, K, Kpad >> 1, N, Npad);
  };
  pack(fe_w1, fe_b1,     w1p,   7,  32,  96,  96);   // bias folded into row 7
  pack(fe_w2, nullptr,   w2p,  96,  96, 160, 160);
  pack(fe_w3, nullptr,   w3p, 160, 160, 192, 192);
  pack(fn_w1, fn_b1,     n1p, 195, 224, 256, 256);   // bias folded into row 195
  pack(fn_w2, nullptr,   n2p, 256, 256, 256, 256);
  pack(fn_w3, nullptr,   n3p, 256, 256,   3,  16);   // N padded 3 -> 16

  mp_edge_kernel<<<128 * 75, 128, 0, stream>>>(
      x, (const v4u*)w1p, (const v4u*)w2p, fe_b2, (const v4u*)w3p, fe_b3, agg);
  mp_node_kernel<<<(128 * 75) / 16, 128, 0, stream>>>(
      agg, x, (const v4u*)n1p, (const v4u*)n2p, fn_b2, (const v4u*)n3p, fn_b3, (float*)d_out);
}
